// BeamSearchDecoder_32478542692728
// MI455X (gfx1250) — matrix operand order repaired and checked by reading, not compiled, not run on hardware
//
#include <hip/hip_runtime.h>

// Shapes from reference
#define BB 32
#define SS 64
#define HH 512
#define EE 256
#define VV 32000
#define KK 3
#define BK 96            // B*K beams
#define G3 1536          // 3*H
#define MAXLEN 32
#define START_TOK 1
#define MTILES 6         // BK/16, same for every GEMM in this model

typedef __attribute__((ext_vector_type(16))) __bf16 v16bf;
typedef __attribute__((ext_vector_type(8)))  float  v8f;

union BF16Frag { v16bf v; unsigned short s[16]; };

__device__ __forceinline__ unsigned short f32_to_bf16(float f) {
    unsigned int u = __float_as_uint(f);
    unsigned int r = 0x7FFFu + ((u >> 16) & 1u);   // round-to-nearest-even
    u += r;
    return (unsigned short)(u >> 16);
}

// ---------------- weight conversion fp32 -> bf16 (once; stays hot in 192MB L2) ----
__global__ void cvt_bf16_kernel(const float* __restrict__ src,
                                unsigned short* __restrict__ dst, int n) {
    int i = blockIdx.x * blockDim.x + threadIdx.x;
    if (i < n) dst[i] = f32_to_bf16(src[i]);
}

// ---------------- state init: h0 = repeat(enc_hidden, K), seqs=START, scores=0 ----
__global__ void init_state_kernel(const float* __restrict__ enc_h,
                                  float* h_cur, float* scores, int* tok, int* seqs) {
    int i = blockIdx.x * blockDim.x + threadIdx.x;
    if (i < BK * HH) {
        int r = i / HH, j = i % HH;
        h_cur[i] = enc_h[(r / KK) * HH + j];
    }
    if (i < BK) { scores[i] = 0.f; tok[i] = START_TOK; }
    if (i < BK * (MAXLEN + 1)) seqs[i] = START_TOK;
}

// ---------------- x = relu(emb[tok]) -> bf16 ; h_cur -> bf16 ---------------------
__global__ void embed_cvt_kernel(const float* __restrict__ emb,
                                 const int* __restrict__ tok,
                                 const float* __restrict__ h_cur,
                                 unsigned short* __restrict__ x_bf,
                                 unsigned short* __restrict__ h_bf) {
    int i = blockIdx.x * blockDim.x + threadIdx.x;
    if (i < BK * EE) {
        int r = i / EE, e = i % EE;
        float v = emb[(size_t)tok[r] * EE + e];
        x_bf[i] = f32_to_bf16(v > 0.f ? v : 0.f);
    } else {
        int i2 = i - BK * EE;
        if (i2 < BK * HH) h_bf[i2] = f32_to_bf16(h_cur[i2]);
    }
}

// ---------------- WMMA bf16 GEMM with register-level B reuse ---------------------
// C[96,N] = A[96,K] * Bt[N,K]^T + bias.
// One wave per 16-wide N strip; the wave holds the B fragment in VGPRs and
// reuses it for all 6 M-tiles (6 WMMAs per fragment load) -> W_out stream read
// exactly once per step (32.8MB, resident in 192MB L2 across all 32 steps).
// All 14 loads of a K-step are issued before the 6 WMMAs so loads overlap math.
__global__ void wmma_gemm_bf16_kernel(const unsigned short* __restrict__ A,
                                      const unsigned short* __restrict__ Bt,
                                      const float* __restrict__ bias,
                                      float* __restrict__ C,
                                      int N, int Kd) {
    int wave = (blockIdx.x * blockDim.x + threadIdx.x) >> 5;
    int lane = threadIdx.x & 31;
    int tilesN = N >> 4;
    if (wave >= tilesN) return;               // wave-uniform exit, EXEC stays full
    int n0 = wave << 4;
    int half = lane >> 4;                     // lane half selects K sub-ranges
    int l    = lane & 15;

    const unsigned short* arow = A  + (size_t)l * Kd;        // row l of M-tile 0
    const unsigned short* brow = Bt + (size_t)(n0 + l) * Kd; // column n0+l of B

    v8f acc[MTILES];
#pragma unroll
    for (int m = 0; m < MTILES; m++) acc[m] = (v8f){};

    for (int k0 = 0; k0 < Kd; k0 += 32) {
        // Phase 1: issue every load of this K-step (independent -> load clause).
        // B frag: column n = n0+l; lanes 0-15 K={0..7,16..23}, lanes 16-31 K={8..15,24..31}
        BF16Frag b;
        const unsigned short* bp = brow + k0 + half * 8;
#pragma unroll
        for (int j = 0; j < 8; j++) { b.s[j] = bp[j]; b.s[8 + j] = bp[16 + j]; }
        // A frags (distinct registers per M-tile):
        // lanes 0-15 K={0..7,16..23}, lanes 16-31 K={8..15,24..31}
        BF16Frag a[MTILES];
#pragma unroll
        for (int m = 0; m < MTILES; m++) {
            const unsigned short* ap = arow + (size_t)(m * 16) * Kd + k0 + half * 8;
#pragma unroll
            for (int j = 0; j < 8; j++) { a[m].s[j] = ap[j]; a[m].s[8 + j] = ap[16 + j]; }
        }
        if (k0 + 32 < Kd) __builtin_prefetch(bp + 32, 0, 1); // next K slice of B
        // Phase 2: 6 WMMAs reusing the B fragment from registers.
#pragma unroll
        for (int m = 0; m < MTILES; m++) {
            acc[m] = __builtin_amdgcn_wmma_f32_16x16x32_bf16(
                /*neg_a=*/false, a[m].v, /*neg_b=*/false, b.v,
                /*c_mod=*/(short)0, acc[m], /*reuse_a=*/false, /*reuse_b=*/false);
        }
    }
    int col = n0 + l;
    float bv = bias ? bias[col] : 0.f;
#pragma unroll
    for (int m = 0; m < MTILES; m++) {
#pragma unroll
        for (int i = 0; i < 8; i++) {
            int row = m * 16 + half * 8 + i;  // C layout: VGPR i -> M = i + 8*half
            C[(size_t)row * N + col] = acc[m][i] + bv;
        }
    }
}

// ---------------- GRU gate combine (biases already added in GEMMs) ---------------
__global__ void gru_combine_kernel(const float* __restrict__ gi,
                                   const float* __restrict__ gh,
                                   const float* __restrict__ h_cur,
                                   float* __restrict__ h_new,
                                   unsigned short* __restrict__ hnew_bf) {
    int i = blockIdx.x * blockDim.x + threadIdx.x;
    if (i >= BK * HH) return;
    int r = i / HH, j = i % HH;
    const float* gir = gi + (size_t)r * G3;
    const float* ghr = gh + (size_t)r * G3;
    float i_r = gir[j], i_z = gir[HH + j], i_n = gir[2 * HH + j];
    float h_r = ghr[j], h_z = ghr[HH + j], h_n = ghr[2 * HH + j];
    float rg = 1.f / (1.f + __expf(-(i_r + h_r)));
    float zg = 1.f / (1.f + __expf(-(i_z + h_z)));
    float ng = tanhf(i_n + rg * h_n);
    float hn = (1.f - zg) * ng + zg * h_cur[i];
    h_new[i] = hn;
    hnew_bf[i] = f32_to_bf16(hn);
}

// ---------------- per-row logsumexp + top-3 over V=32000 ------------------------
__global__ void row_topk_kernel(const float* __restrict__ logits,
                                float* __restrict__ next_scores,
                                int* __restrict__ next_tokens) {
    const int row = blockIdx.x;
    const int tid = threadIdx.x;
    const float* lrow = logits + (size_t)row * VV;
    __shared__ float red[256];
    __shared__ float sv[256 * 3];
    __shared__ int   si[256 * 3];

    // row max
    float m = -3.4e38f;
    for (int c = tid; c < VV; c += 256) { float x = lrow[c]; m = x > m ? x : m; }
    red[tid] = m; __syncthreads();
    for (int s = 128; s > 0; s >>= 1) {
        if (tid < s) { float o = red[tid + s]; if (o > red[tid]) red[tid] = o; }
        __syncthreads();
    }
    float rowmax = red[0]; __syncthreads();

    // sum exp
    float sum = 0.f;
    for (int c = tid; c < VV; c += 256) sum += __expf(lrow[c] - rowmax);
    red[tid] = sum; __syncthreads();
    for (int s = 128; s > 0; s >>= 1) {
        if (tid < s) red[tid] += red[tid + s];
        __syncthreads();
    }
    float lse = rowmax + __logf(red[0]);
    __syncthreads();

    // per-thread top-3 over strided columns (ties -> lowest index)
    float v0 = -3.4e38f, v1 = -3.4e38f, v2 = -3.4e38f; int i0 = 0, i1 = 0, i2 = 0;
    for (int c = tid; c < VV; c += 256) {
        float x = lrow[c];
        if (x > v0)      { v2 = v1; i2 = i1; v1 = v0; i1 = i0; v0 = x; i0 = c; }
        else if (x > v1) { v2 = v1; i2 = i1; v1 = x; i1 = c; }
        else if (x > v2) { v2 = x; i2 = c; }
    }
    sv[tid * 3] = v0; si[tid * 3] = i0;
    sv[tid * 3 + 1] = v1; si[tid * 3 + 1] = i1;
    sv[tid * 3 + 2] = v2; si[tid * 3 + 2] = i2;
    __syncthreads();

    // stage 1: 32 threads each merge 8 candidate lists (compute in regs)
    float w0 = -3.4e38f, w1 = -3.4e38f, w2 = -3.4e38f; int j0 = 0, j1 = 0, j2 = 0;
    if (tid < 32) {
        for (int t2 = tid; t2 < 256; t2 += 32) {
            for (int q = 0; q < 3; q++) {
                float x = sv[t2 * 3 + q]; int ix = si[t2 * 3 + q];
                if (x > w0 || (x == w0 && ix < j0)) { w2 = w1; j2 = j1; w1 = w0; j1 = j0; w0 = x; j0 = ix; }
                else if (x > w1 || (x == w1 && ix < j1)) { w2 = w1; j2 = j1; w1 = x; j1 = ix; }
                else if (x > w2 || (x == w2 && ix < j2)) { w2 = x; j2 = ix; }
            }
        }
    }
    __syncthreads();
    if (tid < 32) {
        sv[tid * 3] = w0; si[tid * 3] = j0;
        sv[tid * 3 + 1] = w1; si[tid * 3 + 1] = j1;
        sv[tid * 3 + 2] = w2; si[tid * 3 + 2] = j2;
    }
    __syncthreads();

    // stage 2: thread 0 merges 96 candidates
    if (tid == 0) {
        float f0 = -3.4e38f, f1 = -3.4e38f, f2 = -3.4e38f; int g0 = 0, g1 = 0, g2 = 0;
        for (int q = 0; q < 96; q++) {
            float x = sv[q]; int ix = si[q];
            if (x > f0 || (x == f0 && ix < g0)) { f2 = f1; g2 = g1; f1 = f0; g1 = g0; f0 = x; g0 = ix; }
            else if (x > f1 || (x == f1 && ix < g1)) { f2 = f1; g2 = g1; f1 = x; g1 = ix; }
            else if (x > f2 || (x == f2 && ix < g2)) { f2 = x; g2 = ix; }
        }
        next_scores[row * 3]     = f0 - lse; next_tokens[row * 3]     = g0;
        next_scores[row * 3 + 1] = f1 - lse; next_tokens[row * 3 + 1] = g1;
        next_scores[row * 3 + 2] = f2 - lse; next_tokens[row * 3 + 2] = g2;
    }
}

// ---------------- per-batch beam merge: top-3 of 3x3, stable tie-break -----------
__global__ void beam_update_kernel(float* __restrict__ scores,
                                   const float* __restrict__ next_scores,
                                   const int* __restrict__ next_tokens,
                                   int* __restrict__ tok, int* __restrict__ src) {
    int b = threadIdx.x;
    if (b >= BB) return;
    float tot[9];
    for (int k = 0; k < KK; k++) {
        float s = scores[b * KK + k];
        for (int q = 0; q < KK; q++) tot[k * KK + q] = s + next_scores[(b * KK + k) * KK + q];
    }
    bool used[9] = {false, false, false, false, false, false, false, false, false};
    float ns[3]; int nt[3]; int sb[3];
    for (int j = 0; j < KK; j++) {
        float best = -3.4e38f; int bi = 0;
        for (int i = 0; i < 9; i++)
            if (!used[i] && tot[i] > best) { best = tot[i]; bi = i; }
        used[bi] = true;
        int beam = bi / KK, tki = bi % KK;
        ns[j] = best;
        sb[j] = b * KK + beam;
        nt[j] = next_tokens[(b * KK + beam) * KK + tki];
    }
    for (int j = 0; j < KK; j++) {
        scores[b * KK + j] = ns[j];
        tok[b * KK + j]    = nt[j];
        src[b * KK + j]    = sb[j];
    }
}

// ---------------- reorder h by beam, gather+extend sequences ---------------------
__global__ void gather_state_kernel(const int* __restrict__ src,
                                    const float* __restrict__ h_new,
                                    float* __restrict__ h_cur,
                                    const int* __restrict__ seqs_in,
                                    int* __restrict__ seqs_out,
                                    const int* __restrict__ tok, int t) {
    int r = blockIdx.x;
    int j = threadIdx.x;
    int s = src[r];
    h_cur[(size_t)r * HH + j] = h_new[(size_t)s * HH + j];
    if (j < MAXLEN + 1) {
        int v = seqs_in[s * (MAXLEN + 1) + j];
        seqs_out[r * (MAXLEN + 1) + j] = (j == t + 1) ? tok[r] : v;
    }
}

// ---------------- output: zero-fill then sparse one-hot + h + scores -------------
__global__ void zero_out_kernel(float* __restrict__ out, size_t n) {
    size_t i = (size_t)blockIdx.x * blockDim.x + threadIdx.x;
    size_t stride = (size_t)gridDim.x * blockDim.x;
    for (; i < n; i += stride) out[i] = 0.f;
}

__global__ void finalize_kernel(const int* __restrict__ seqs,
                                const float* __restrict__ h_cur,
                                const float* __restrict__ scores,
                                float* __restrict__ out) {
    int i = blockIdx.x * blockDim.x + threadIdx.x;
    const size_t OFF_H = (size_t)BB * (MAXLEN + 1) * VV;
    const size_t OFF_S = OFF_H + (size_t)BK * HH;
    if (i < BB * (MAXLEN + 1)) {
        int b = i / (MAXLEN + 1), l = i % (MAXLEN + 1);
        int v = seqs[(b * KK) * (MAXLEN + 1) + l];     // beam 0 is best
        out[(size_t)i * VV + v] = 1.0f;
    }
    if (i < BK * HH) out[OFF_H + i] = h_cur[i];
    if (i < BK)      out[OFF_S + i] = scores[i];
}

extern "C" void kernel_launch(void* const* d_in, const int* in_sizes, int n_in,
                              void* d_out, int out_size, void* d_ws, size_t ws_size,
                              hipStream_t stream) {
    (void)in_sizes; (void)n_in; (void)ws_size;
    // Inputs in setup_inputs() order:
    // 0 encoder_outputs (unused), 1 encoder_hidden, 2 emb, 3 W_ih, 4 W_hh,
    // 5 b_ih, 6 b_hh, 7 W_out, 8 b_out
    const float* enc_h = (const float*)d_in[1];
    const float* emb   = (const float*)d_in[2];
    const float* W_ih  = (const float*)d_in[3];
    const float* W_hh  = (const float*)d_in[4];
    const float* b_ih  = (const float*)d_in[5];
    const float* b_hh  = (const float*)d_in[6];
    const float* W_out = (const float*)d_in[7];
    const float* b_out = (const float*)d_in[8];
    float* out = (float*)d_out;

    // Workspace carve-out (256B aligned)
    char* base = (char*)d_ws;
    size_t off = 0;
    auto alloc = [&](size_t bytes) -> void* {
        void* r = base + off;
        off = (off + bytes + 255) & ~(size_t)255;
        return r;
    };
    unsigned short* wih_bf  = (unsigned short*)alloc((size_t)G3 * EE * 2);
    unsigned short* whh_bf  = (unsigned short*)alloc((size_t)G3 * HH * 2);
    unsigned short* wout_bf = (unsigned short*)alloc((size_t)VV * HH * 2);
    unsigned short* x_bf    = (unsigned short*)alloc((size_t)BK * EE * 2);
    unsigned short* h_bf    = (unsigned short*)alloc((size_t)BK * HH * 2);
    unsigned short* hnew_bf = (unsigned short*)alloc((size_t)BK * HH * 2);
    float* gi      = (float*)alloc((size_t)BK * G3 * 4);
    float* gh      = (float*)alloc((size_t)BK * G3 * 4);
    float* h_cur   = (float*)alloc((size_t)BK * HH * 4);
    float* h_new   = (float*)alloc((size_t)BK * HH * 4);
    float* logits  = (float*)alloc((size_t)BK * VV * 4);
    float* nsc     = (float*)alloc((size_t)BK * KK * 4);
    int*   ntk     = (int*)alloc((size_t)BK * KK * 4);
    float* scores  = (float*)alloc((size_t)BK * 4);
    int*   tok     = (int*)alloc((size_t)BK * 4);
    int*   src     = (int*)alloc((size_t)BK * 4);
    int*   seqsA   = (int*)alloc((size_t)BK * (MAXLEN + 1) * 4);
    int*   seqsB   = (int*)alloc((size_t)BK * (MAXLEN + 1) * 4);

    // One-time weight conversion (idempotent; W_out bf16 = 32.8MB stays in L2)
    {
        int n;
        n = G3 * EE; cvt_bf16_kernel<<<(n + 255) / 256, 256, 0, stream>>>(W_ih, wih_bf, n);
        n = G3 * HH; cvt_bf16_kernel<<<(n + 255) / 256, 256, 0, stream>>>(W_hh, whh_bf, n);
        n = VV * HH; cvt_bf16_kernel<<<(n + 255) / 256, 256, 0, stream>>>(W_out, wout_bf, n);
    }
    init_state_kernel<<<(BK * HH + 255) / 256, 256, 0, stream>>>(enc_h, h_cur, scores, tok, seqsA);

    int* seqs_in = seqsA;
    int* seqs_out = seqsB;
    for (int t = 0; t < MAXLEN; t++) {
        embed_cvt_kernel<<<(BK * EE + BK * HH + 255) / 256, 256, 0, stream>>>(
            emb, tok, h_cur, x_bf, h_bf);

        // gi = relu(emb) @ W_ih^T + b_ih  (N=1536,K=256): 96 waves, B reused 6x
        {
            int waves = G3 / 16;
            wmma_gemm_bf16_kernel<<<(waves * 32 + 255) / 256, 256, 0, stream>>>(
                x_bf, wih_bf, b_ih, gi, G3, EE);
        }
        // gh = h @ W_hh^T + b_hh  (N=1536,K=512)
        {
            int waves = G3 / 16;
            wmma_gemm_bf16_kernel<<<(waves * 32 + 255) / 256, 256, 0, stream>>>(
                h_bf, whh_bf, b_hh, gh, G3, HH);
        }
        gru_combine_kernel<<<(BK * HH + 255) / 256, 256, 0, stream>>>(
            gi, gh, h_cur, h_new, hnew_bf);

        // logits = h_new @ W_out^T + b_out  (N=32000,K=512): 2000 waves, B reused 6x
        {
            int waves = VV / 16;
            wmma_gemm_bf16_kernel<<<(waves * 32 + 255) / 256, 256, 0, stream>>>(
                hnew_bf, wout_bf, b_out, logits, VV, HH);
        }
        row_topk_kernel<<<BK, 256, 0, stream>>>(logits, nsc, ntk);
        beam_update_kernel<<<1, 32, 0, stream>>>(scores, nsc, ntk, tok, src);
        gather_state_kernel<<<BK, HH, 0, stream>>>(src, h_new, h_cur, seqs_in, seqs_out, tok, t);
        int* tmp = seqs_in; seqs_in = seqs_out; seqs_out = tmp;
    }
    // After 32 steps (even number of swaps) final state is in seqs_in.
    zero_out_kernel<<<8192, 256, 0, stream>>>(out, (size_t)out_size);
    finalize_kernel<<<(BK * HH + 255) / 256, 256, 0, stream>>>(seqs_in, h_cur, scores, out);
}
